// TemporalLePEAttention_72971494359720
// MI455X (gfx1250) — matrix operand order repaired and checked by reading, not compiled, not run on hardware
//
#include <hip/hip_runtime.h>
#include <hip/hip_bf16.h>
#include <stdint.h>

// ---------------- static problem config (matches reference) ----------------
#define B_    1
#define T_    4
#define H_    32
#define W_    64
#define C_    128
#define HEADS_ 4
#define HD_   32
#define SPLIT_ 8
#define H_SP  32
#define W_SP  8
#define NWW_  8
#define NTOK_LOCAL 1024          // T*H_SP*W_SP
#define NTOK_ALL   2048          // local + pooled (incl. zero tokens)
#define NQKV  ((size_t)T_ * H_ * W_ * C_)   // 1,048,576 floats per q/k/v
#define SCALE_ 0.17677669529663687f          // 32^-0.5

typedef __attribute__((ext_vector_type(16))) _Float16 v16h;
typedef __attribute__((ext_vector_type(8)))  float    v8f;
typedef __attribute__((ext_vector_type(4)))  unsigned int u32x4;
typedef __attribute__((ext_vector_type(4)))  float    f32x4;

union U16h { v16h v; u32x4 q[2]; _Float16 h[16]; };

// ---------------- workspace layout (bytes) ----------------
#define WS_PMK   0u                           // 4*128*8*8 f32   = 131072 B
#define WS_PMV   131072u                      // 131072 B
#define WS_LEPE  262144u                      // 8*4*256*128 f32 = 4,194,304 B
#define WS_QS    4456448u                     // 32*1024*32 f16  = 2,097,152 B
#define WS_KALL  6553600u                     // 32*2048*32 f16  = 4,194,304 B
#define WS_VALLT 10747904u                    // 32*32*2048 f16  = 4,194,304 B

// ---------------- DPP cross-lane helpers (16-lane-half reductions) ---------
// dpp8 selector packs: sel[i] at bits [3i+2:3i]
#define DPP8_XOR1 (1 | (0<<3) | (3<<6) | (2<<9) | (5<<12) | (4<<15) | (7<<18) | (6<<21))
#define DPP8_XOR2 (2 | (3<<3) | (0<<6) | (1<<9) | (6<<12) | (7<<15) | (4<<18) | (5<<21))
#define DPP8_XOR4 (4 | (5<<3) | (6<<6) | (7<<9) | (0<<12) | (1<<15) | (2<<18) | (3<<21))
#define DPP16_ROW_ROR8 0x128    // row_ror:8 -> lane^8 within each 16-lane row

__device__ __forceinline__ float dpp8_xor1(float x) {
    return __builtin_bit_cast(float,
        __builtin_amdgcn_mov_dpp8(__builtin_bit_cast(int, x), DPP8_XOR1));
}
__device__ __forceinline__ float dpp8_xor2(float x) {
    return __builtin_bit_cast(float,
        __builtin_amdgcn_mov_dpp8(__builtin_bit_cast(int, x), DPP8_XOR2));
}
__device__ __forceinline__ float dpp8_xor4(float x) {
    return __builtin_bit_cast(float,
        __builtin_amdgcn_mov_dpp8(__builtin_bit_cast(int, x), DPP8_XOR4));
}
__device__ __forceinline__ float dpp_ror8(float x) {
    return __builtin_bit_cast(float,
        __builtin_amdgcn_update_dpp(0, __builtin_bit_cast(int, x),
                                    DPP16_ROW_ROR8, 0xF, 0xF, true));
}
// max/sum across the 16-lane half this lane belongs to (result in all lanes)
__device__ __forceinline__ float half16_max(float x) {
    x = fmaxf(x, dpp8_xor1(x));
    x = fmaxf(x, dpp8_xor2(x));
    x = fmaxf(x, dpp8_xor4(x));
    x = fmaxf(x, dpp_ror8(x));
    return x;
}
__device__ __forceinline__ float half16_sum(float x) {
    x += dpp8_xor1(x);
    x += dpp8_xor2(x);
    x += dpp8_xor4(x);
    x += dpp_ror8(x);
    return x;
}

// =====================================================================
// Stage A: window pooling  pm[t][c][s][wcol] = sum_n win(x)[c][n]*pool_w[s][n]+b[s]
// =====================================================================
__global__ __launch_bounds__(1024) void pool_kernel(
    const float* __restrict__ qkv, const float* __restrict__ pool_w,
    const float* __restrict__ pool_b, float* __restrict__ pm_k,
    float* __restrict__ pm_v)
{
    const int wcol = blockIdx.x;        // window column 0..7
    const int t    = blockIdx.y;        // time 0..3
    const int kv   = blockIdx.z;        // 0 = k, 1 = v
    const float* src = qkv + (size_t)(1 + kv) * NQKV;
    float* pm = kv ? pm_v : pm_k;

    const int s = threadIdx.x >> 7;     // split 0..7
    const int c = threadIdx.x & 127;    // channel 0..127

    float acc = pool_b[s];
    #pragma unroll 4
    for (int n = 0; n < 256; ++n) {
        const int hs = n >> 3, wsp = n & 7;
        const float x = src[(((size_t)t * H_ + hs) * W_ + (wcol * W_SP + wsp)) * C_ + c];
        acc += x * pool_w[s * 256 + n];
    }
    pm[(((size_t)t * C_ + c) * SPLIT_ + s) * NWW_ + wcol] = acc;
}

// =====================================================================
// Stage B: pack Q (scaled) / K_all / V_all^T into fp16 token layouts
// =====================================================================
__global__ __launch_bounds__(256) void pack_kernel(
    const float* __restrict__ qkv, const float* __restrict__ pm_k,
    const float* __restrict__ pm_v, _Float16* __restrict__ Qs,
    _Float16* __restrict__ Kall, _Float16* __restrict__ VallT)
{
    const int idx = blockIdx.x * 256 + threadIdx.x;
    if (idx >= 163840) return;

    int tensor, wh, n;
    if (idx < 32768)       { tensor = 0; wh = idx >> 10;             n = idx & 1023; }
    else if (idx < 98304)  { tensor = 1; int r = idx - 32768; wh = r >> 11; n = r & 2047; }
    else                   { tensor = 2; int r = idx - 98304; wh = r >> 11; n = r & 2047; }
    const int win = wh >> 2, head = wh & 3;

    float vals[32];
    const float* src = qkv + (size_t)tensor * NQKV;

    if (n < NTOK_LOCAL) {
        const int t = n >> 8, hs = (n >> 3) & 31, wsp = n & 7;
        const f32x4* p = (const f32x4*)(src +
            (((size_t)t * H_ + hs) * W_ + (win * W_SP + wsp)) * C_ + head * HD_);
        #pragma unroll
        for (int i = 0; i < 8; ++i) {
            f32x4 f = p[i];
            vals[4*i+0] = f.x; vals[4*i+1] = f.y; vals[4*i+2] = f.z; vals[4*i+3] = f.w;
        }
    } else {
        // focal pooled token: n_p = t*256 + kh*32 + kwcol ; col = win + kwcol - 16
        const float* pm = (tensor == 1) ? pm_k : pm_v;
        const int np = n - NTOK_LOCAL;
        const int t = np >> 8, kh = (np >> 5) & 7, kwcol = np & 31;
        const int col = win + kwcol - 16;
        if (col >= 0 && col < NWW_) {
            #pragma unroll
            for (int d = 0; d < 32; ++d)
                vals[d] = pm[(((size_t)t * C_ + head * HD_ + d) * SPLIT_ + kh) * NWW_ + col];
        } else {
            #pragma unroll
            for (int d = 0; d < 32; ++d) vals[d] = 0.0f;
        }
    }

    if (tensor == 0) {
        alignas(16) _Float16 tmp[32];
        #pragma unroll
        for (int d = 0; d < 32; ++d) tmp[d] = (_Float16)(vals[d] * SCALE_);
        u32x4* dst = (u32x4*)(Qs + ((size_t)wh * NTOK_LOCAL + n) * HD_);
        const u32x4* s4 = (const u32x4*)tmp;
        dst[0] = s4[0]; dst[1] = s4[1]; dst[2] = s4[2]; dst[3] = s4[3];
    } else if (tensor == 1) {
        alignas(16) _Float16 tmp[32];
        #pragma unroll
        for (int d = 0; d < 32; ++d) tmp[d] = (_Float16)vals[d];
        u32x4* dst = (u32x4*)(Kall + ((size_t)wh * NTOK_ALL + n) * HD_);
        const u32x4* s4 = (const u32x4*)tmp;
        dst[0] = s4[0]; dst[1] = s4[1]; dst[2] = s4[2]; dst[3] = s4[3];
    } else {
        #pragma unroll
        for (int d = 0; d < 32; ++d)
            VallT[((size_t)wh * HD_ + d) * NTOK_ALL + n] = (_Float16)vals[d];
    }
}

// =====================================================================
// Stage C: LePE depthwise 3x3 per strip window (zero pad at window edges)
// lepe[((win*T + t)*256 + hs*8 + wsp)*128 + c]
// =====================================================================
__global__ __launch_bounds__(256) void lepe_kernel(
    const float* __restrict__ qkv, const float* __restrict__ conv_w,
    const float* __restrict__ conv_b, float* __restrict__ lepe)
{
    const int idx = blockIdx.x * 256 + threadIdx.x;     // 1,048,576 total
    const int c   = idx & 127;
    const int pos = idx >> 7;                            // ((win*4+t)*256 + hs*8+wsp)
    const int wsp = pos & 7, hs = (pos >> 3) & 31, t = (pos >> 8) & 3, win = pos >> 10;
    const float* v = qkv + 2 * NQKV;

    float acc = conv_b[c];
    #pragma unroll
    for (int ky = 0; ky < 3; ++ky) {
        const int y = hs + ky - 1;
        if (y < 0 || y >= H_SP) continue;
        #pragma unroll
        for (int kx = 0; kx < 3; ++kx) {
            const int x = wsp + kx - 1;
            if (x < 0 || x >= W_SP) continue;
            acc += v[(((size_t)t * H_ + y) * W_ + (win * W_SP + x)) * C_ + c]
                 * conv_w[c * 9 + ky * 3 + kx];
        }
    }
    lepe[(size_t)pos * C_ + c] = acc;
}

// =====================================================================
// Stage D: flash attention per (window, head, q-chunk). 8 waves/block,
// each wave owns 16 q rows. 64 chunks of 32 keys: 2 QK^T + 2 PV WMMAs.
// Softmax row reductions are pure-VALU DPP (no LDS round trips).
// =====================================================================
__global__ __launch_bounds__(256) void attn_kernel(
    const _Float16* __restrict__ Qs, const _Float16* __restrict__ Kall,
    const _Float16* __restrict__ VallT, const float* __restrict__ lepe,
    float* __restrict__ out)
{
    __shared__ _Float16 lds_p[8][16][32];   // per-wave P re-layout tile (8 KB)

    const int tid  = threadIdx.x;
    const int wave = tid >> 5, lane = tid & 31;
    const int h2 = lane >> 4, ln = lane & 15;

    const int blk    = blockIdx.x;          // 0..255
    const int qchunk = blk & 7;
    const int head   = (blk >> 3) & 3;
    const int win    = blk >> 5;
    const int wh     = win * HEADS_ + head;
    const int qtok0  = qchunk * 128 + wave * 16;

    const _Float16* qbase  = Qs    + (size_t)wh * NTOK_LOCAL * HD_;
    const _Float16* kbase  = Kall  + (size_t)wh * NTOK_ALL   * HD_;
    const _Float16* vtbase = VallT + (size_t)wh * HD_ * NTOK_ALL;

    // ---- Q A-fragment (16x32): lane row = ln; elem e -> K=16*h2 + e
    U16h qf;
    {
        const _Float16* qrow = qbase + (size_t)(qtok0 + ln) * HD_ + h2 * 16;
        qf.q[0] = *(const u32x4*)(qrow);
        qf.q[1] = *(const u32x4*)(qrow + 8);
    }

    float rm[8], rl[8];
    v8f o0 = {}, o1 = {};
    #pragma unroll
    for (int r = 0; r < 8; ++r) { rm[r] = -1e30f; rl[r] = 0.0f; }

    for (int kc = 0; kc < 64; ++kc) {
        const int c0 = kc * 32;
        if (kc + 1 < 64) {
            __builtin_prefetch(kbase  + (size_t)(c0 + 32) * HD_, 0, 3);
            __builtin_prefetch(vtbase + (size_t)ln * NTOK_ALL + c0 + 32, 0, 3);
        }

        // ---- K^T B-fragments: col = key (ln), elem e -> dim = 16*h2 + e
        U16h b0, b1;
        {
            const _Float16* kp0 = kbase + (size_t)(c0 + ln) * HD_ + h2 * 16;
            b0.q[0] = *(const u32x4*)kp0;       b0.q[1] = *(const u32x4*)(kp0 + 8);
            const _Float16* kp1 = kbase + (size_t)(c0 + 16 + ln) * HD_ + h2 * 16;
            b1.q[0] = *(const u32x4*)kp1;       b1.q[1] = *(const u32x4*)(kp1 + 8);
        }
        const v8f zero = {};
        v8f s0 = __builtin_amdgcn_wmma_f32_16x16x32_f16(false, qf.v, false, b0.v,
                                                        (short)0, zero, false, false);
        v8f s1 = __builtin_amdgcn_wmma_f32_16x16x32_f16(false, qf.v, false, b1.v,
                                                        (short)0, zero, false, false);

        // ---- online softmax over this 32-key chunk (rows live in 16-lane halves)
        float p0[8], p1[8];
        #pragma unroll
        for (int r = 0; r < 8; ++r) {
            const float mx   = half16_max(fmaxf(s0[r], s1[r]));
            const float mnew  = fmaxf(rm[r], mx);
            const float alpha = __expf(rm[r] - mnew);
            rm[r] = mnew;
            const float e0 = __expf(s0[r] - mnew);
            const float e1 = __expf(s1[r] - mnew);
            const float rs = half16_sum(e0 + e1);
            rl[r] = rl[r] * alpha + rs;
            o0[r] *= alpha;
            o1[r] *= alpha;
            p0[r] = e0; p1[r] = e1;
        }

        // ---- C-layout P -> A-layout via per-wave LDS tile
        #pragma unroll
        for (int r = 0; r < 8; ++r) {
            lds_p[wave][r + 8 * h2][ln]      = (_Float16)p0[r];
            lds_p[wave][r + 8 * h2][16 + ln] = (_Float16)p1[r];
        }
        asm volatile("s_wait_dscnt 0" ::: "memory");   // same-wave LDS RAW turnaround
        U16h pf;
        pf.q[0] = *(const u32x4*)&lds_p[wave][ln][16 * h2];
        pf.q[1] = *(const u32x4*)&lds_p[wave][ln][16 * h2 + 8];
        asm volatile("" ::: "memory");

        // ---- V B-fragments from transposed V: col = dim, elem e -> key = c0+16*h2+e
        U16h bv0, bv1;
        {
            const _Float16* vp0 = vtbase + (size_t)ln * NTOK_ALL + c0 + h2 * 16;
            bv0.q[0] = *(const u32x4*)vp0;      bv0.q[1] = *(const u32x4*)(vp0 + 8);
            const _Float16* vp1 = vtbase + (size_t)(16 + ln) * NTOK_ALL + c0 + h2 * 16;
            bv1.q[0] = *(const u32x4*)vp1;      bv1.q[1] = *(const u32x4*)(vp1 + 8);
        }
        o0 = __builtin_amdgcn_wmma_f32_16x16x32_f16(false, pf.v, false, bv0.v,
                                                    (short)0, o0, false, false);
        o1 = __builtin_amdgcn_wmma_f32_16x16x32_f16(false, pf.v, false, bv1.v,
                                                    (short)0, o1, false, false);
    }

    // ---- epilogue: normalize, add LePE, scatter to (B,T,H,W,C)
    #pragma unroll
    for (int r = 0; r < 8; ++r) {
        const float inv  = 1.0f / rl[r];
        const int token = qtok0 + r + 8 * h2;
        const int t = token >> 8, hs = (token >> 3) & 31, wsp = token & 7;
        const size_t obase = (((size_t)t * H_ + hs) * W_ + (win * W_SP + wsp)) * C_ + head * HD_;
        const size_t lbase = (((size_t)(win * T_ + t) * 256) + hs * 8 + wsp) * C_ + head * HD_;
        out[obase + ln]      = o0[r] * inv + lepe[lbase + ln];
        out[obase + 16 + ln] = o1[r] * inv + lepe[lbase + 16 + ln];
    }
}

// =====================================================================
extern "C" void kernel_launch(void* const* d_in, const int* in_sizes, int n_in,
                              void* d_out, int out_size, void* d_ws, size_t ws_size,
                              hipStream_t stream)
{
    const float* qkv    = (const float*)d_in[0];
    const float* conv_w = (const float*)d_in[1];
    const float* conv_b = (const float*)d_in[2];
    const float* pool_w = (const float*)d_in[3];
    const float* pool_b = (const float*)d_in[4];
    float* out = (float*)d_out;

    char* ws = (char*)d_ws;
    float*    pm_k  = (float*)(ws + WS_PMK);
    float*    pm_v  = (float*)(ws + WS_PMV);
    float*    lepe  = (float*)(ws + WS_LEPE);
    _Float16* Qs    = (_Float16*)(ws + WS_QS);
    _Float16* Kall  = (_Float16*)(ws + WS_KALL);
    _Float16* VallT = (_Float16*)(ws + WS_VALLT);

    pool_kernel<<<dim3(NWW_, T_, 2), 1024, 0, stream>>>(qkv, pool_w, pool_b, pm_k, pm_v);
    pack_kernel<<<640, 256, 0, stream>>>(qkv, pm_k, pm_v, Qs, Kall, VallT);
    lepe_kernel<<<4096, 256, 0, stream>>>(qkv, conv_w, conv_b, lepe);
    attn_kernel<<<256, 256, 0, stream>>>(Qs, Kall, VallT, lepe, out);
}
